// MassConservingLSTM_58772332478725
// MI455X (gfx1250) — matrix operand order repaired and checked by reading, not compiled, not run on hardware
//
#include <hip/hip_runtime.h>
#include <hip/hip_bf16.h>

// ---------------------------------------------------------------------------
// Mass-conserving LSTM scan for MI455X (gfx1250).
// Dominant cost: per-step gate GEMMs (256x160)@(160x16384) and (160x8192),
// 2.1 TFLOP total -> compute bound. Use v_wmma_f32_16x16x32_bf16 (K=32/issue,
// 8x the fp32 WMMA K-depth), f32 accumulate, max-subtracted softmax in f32.
// bf16 weights (7.9 MB) stay resident in the 192 MB L2 for all 1024 steps.
// ---------------------------------------------------------------------------

#define L_STEPS  1024
#define BATCH    256
#define IN_DIM   64
#define AUX_DIM  32
#define OUT_DIM  128
#define GATE_DIM 160   // AUX_DIM + OUT_DIM

typedef __bf16 bf16_t;
typedef bf16_t v16bf __attribute__((ext_vector_type(16)));
typedef bf16_t v8bf  __attribute__((ext_vector_type(8)));
typedef float  v8f   __attribute__((ext_vector_type(8)));

// ---------------------------------------------------------------------------
// fp32 -> bf16 weight conversion (runs once per launch; deterministic)
// ---------------------------------------------------------------------------
__global__ void cvt_bf16_kernel(const float* __restrict__ src,
                                bf16_t* __restrict__ dst, int n) {
  int i = blockIdx.x * blockDim.x + threadIdx.x;
  if (i < n) dst[i] = (bf16_t)src[i];
}

// ---------------------------------------------------------------------------
// Fused state update: finalize step t_done (h/c outputs, new carry), then
// prepare step t_done+1 (c_sum, bf16 gate vector g, o-gate, zero accumulator).
// One block per batch element, 128 threads (= OUT_DIM).
// ---------------------------------------------------------------------------
__global__ void state_kernel(const float* __restrict__ xa,      // (L,B,32)
                             const float* __restrict__ W_o,     // (128,160)
                             const float* __restrict__ b_o,     // (128)
                             float* __restrict__ c_state,       // (B,128)
                             float* __restrict__ c_acc,         // (B,128)
                             bf16_t* __restrict__ g_bf,         // (B,160)
                             float* __restrict__ o_gate,        // (B,128)
                             float* __restrict__ out_h,         // (L,B,128)
                             float* __restrict__ out_c,         // (L,B,128)
                             int t_done) {
  const int b   = blockIdx.x;
  const int tid = threadIdx.x;            // 0..127
  const int idx = b * OUT_DIM + tid;

  __shared__ float g_s[GATE_DIM];
  __shared__ float red[OUT_DIM];

  float c_next;
  if (t_done >= 0) {
    float c_new = c_acc[idx];
    float o     = o_gate[idx];
    float h     = o * c_new;
    out_h[(size_t)t_done * BATCH * OUT_DIM + idx] = h;
    c_next = c_new - h;
    out_c[(size_t)t_done * BATCH * OUT_DIM + idx] = c_next;
  } else {
    c_next = 0.0f;                        // initial carry
  }
  c_state[idx] = c_next;

  const int t = t_done + 1;
  if (t >= L_STEPS) return;

  // --- c_sum reduction over OUT_DIM ---
  red[tid] = c_next;
  __syncthreads();
  for (int s = 64; s > 0; s >>= 1) {
    if (tid < s) red[tid] += red[tid + s];
    __syncthreads();
  }
  const float csum  = red[0];
  const float scale = (csum == 0.0f) ? 1.0f : csum;

  // --- build gate vector g = [xa_t, c/scale] ---
  if (tid < AUX_DIM)
    g_s[tid] = xa[(size_t)t * BATCH * AUX_DIM + (size_t)b * AUX_DIM + tid];
  g_s[AUX_DIM + tid] = c_next / scale;
  __syncthreads();

  bf16_t* grow = g_bf + (size_t)b * GATE_DIM;
  grow[tid] = (bf16_t)g_s[tid];
  if (tid + 128 < GATE_DIM) grow[tid + 128] = (bf16_t)g_s[tid + 128];

  // --- tiny o-gate: sigmoid(g . W_o[tid,:] + b_o[tid]) ---
  float acc = b_o[tid];
  const float* wrow = W_o + (size_t)tid * GATE_DIM;
#pragma unroll 8
  for (int k = 0; k < GATE_DIM; ++k) acc += g_s[k] * wrow[k];
  o_gate[idx] = 1.0f / (1.0f + __expf(-acc));

  // --- zero the c_new accumulator for the coming step ---
  c_acc[idx] = 0.0f;
}

// ---------------------------------------------------------------------------
// Gate GEMM + softmax + weighted scatter.
//   IS_R: logits(16b x 128p) = g_tile @ W[row*128..+128,:]^T, softmax over p,
//         scale by c_state[b,row], atomicAdd into c_acc[b,p].
//  !IS_R: same with W_i, weight = xm[t,b,row].
// Block: 128 threads (4 waves). Wave w owns N-tiles 2w, 2w+1.
// 40 v_wmma_f32_16x16x32_bf16 per block (8 N-tiles x 5 K-chunks).
// ---------------------------------------------------------------------------
template <int N_ROWS, bool IS_R>
__global__ void gate_gemm_kernel(const bf16_t* __restrict__ g_bf,    // (B,160)
                                 const bf16_t* __restrict__ Wb,      // (N_ROWS*128,160)
                                 const float* __restrict__ bias,     // (N_ROWS*128)
                                 const float* __restrict__ c_state,  // (B,128)
                                 const float* __restrict__ xm,       // (L,B,64)
                                 float* __restrict__ c_acc,          // (B,128)
                                 int t) {
  const int row  = blockIdx.x;          // gate row: o (R) or i (I)
  const int b0   = blockIdx.y * 16;     // batch tile base
  const int tid  = threadIdx.x;
  const int wave = tid >> 5;
  const int lane = tid & 31;

  __shared__ float tile[16][OUT_DIM];   // logits, 8 KB

  // --- A fragment addressing (16x32 bf16, ISA layout) ---
  const int m     = lane & 15;
  const int akoff = (lane >> 4) * 8;    // lanes 16-31 start at K=8 / K=24
  const bf16_t* arow = g_bf + (size_t)(b0 + m) * GATE_DIM;

  // --- B fragment addressing (32x16 bf16, same K order as A) ---
  const int nl = lane & 15;
  const int bk = (lane >> 4) * 8;
  const int n0 = wave * 32 + nl;        // N-tile 2w
  const int n1 = n0 + 16;               // N-tile 2w+1
  const bf16_t* bp0 = Wb + ((size_t)row * OUT_DIM + n0) * GATE_DIM + bk;
  const bf16_t* bp1 = Wb + ((size_t)row * OUT_DIM + n1) * GATE_DIM + bk;

  v8f acc0 = {};
  v8f acc1 = {};
#pragma unroll
  for (int kc = 0; kc < 5; ++kc) {      // GATE_DIM = 5 * 32
    const int kbase = kc * 32;
    v8bf alo = *(const v8bf*)(arow + kbase + akoff);        // K  +0..7
    v8bf ahi = *(const v8bf*)(arow + kbase + akoff + 16);   // K +16..23
    v16bf a;
#pragma unroll
    for (int e = 0; e < 8; ++e) { a[e] = alo[e]; a[8 + e] = ahi[e]; }
    v8bf b0lo = *(const v8bf*)(bp0 + kbase);
    v8bf b0hi = *(const v8bf*)(bp0 + kbase + 16);
    v8bf b1lo = *(const v8bf*)(bp1 + kbase);
    v8bf b1hi = *(const v8bf*)(bp1 + kbase + 16);
    v16bf bm0, bm1;
#pragma unroll
    for (int e = 0; e < 8; ++e) { bm0[e] = b0lo[e]; bm0[8 + e] = b0hi[e]; bm1[e] = b1lo[e]; bm1[8 + e] = b1hi[e]; }
    acc0 = __builtin_amdgcn_wmma_f32_16x16x32_bf16(false, a, false, bm0,
                                                   (short)0, acc0, false, false);
    acc1 = __builtin_amdgcn_wmma_f32_16x16x32_bf16(false, a, false, bm1,
                                                   (short)0, acc1, false, false);
  }

  // --- spill accumulators to LDS with bias (C layout: M = r + 8*(lane>=16)) ---
  const int   mofs  = (lane >> 4) * 8;
  const float bias0 = bias[row * OUT_DIM + wave * 32 + nl];
  const float bias1 = bias[row * OUT_DIM + wave * 32 + 16 + nl];
#pragma unroll
  for (int r = 0; r < 8; ++r) {
    tile[r + mofs][wave * 32 + nl]      = acc0[r] + bias0;
    tile[r + mofs][wave * 32 + 16 + nl] = acc1[r] + bias1;
  }
  __syncthreads();

  // --- per-row softmax (8 lanes per row, shfl reductions) + scatter ---
  const int srow = tid >> 3;            // local batch row 0..15
  const int sub  = tid & 7;
  float mx = -1e30f;
  for (int c = sub; c < OUT_DIM; c += 8) mx = fmaxf(mx, tile[srow][c]);
  mx = fmaxf(mx, __shfl_xor(mx, 1, 32));
  mx = fmaxf(mx, __shfl_xor(mx, 2, 32));
  mx = fmaxf(mx, __shfl_xor(mx, 4, 32));
  float sum = 0.0f;
  for (int c = sub; c < OUT_DIM; c += 8) {
    float e = __expf(tile[srow][c] - mx);
    tile[srow][c] = e;
    sum += e;
  }
  sum += __shfl_xor(sum, 1, 32);
  sum += __shfl_xor(sum, 2, 32);
  sum += __shfl_xor(sum, 4, 32);

  float wgt;
  if (IS_R)
    wgt = c_state[(size_t)(b0 + srow) * OUT_DIM + row];
  else
    wgt = xm[(size_t)t * BATCH * IN_DIM + (size_t)(b0 + srow) * IN_DIM + row];
  const float f = wgt / sum;

  float* cr = c_acc + (size_t)(b0 + srow) * OUT_DIM;
  for (int c = sub; c < OUT_DIM; c += 8)
    atomicAdd(cr + c, tile[srow][c] * f);
}

// ---------------------------------------------------------------------------
// Launch
// ---------------------------------------------------------------------------
extern "C" void kernel_launch(void* const* d_in, const int* in_sizes, int n_in,
                              void* d_out, int out_size, void* d_ws, size_t ws_size,
                              hipStream_t stream) {
  (void)in_sizes; (void)n_in; (void)out_size; (void)ws_size;
  const float* xm  = (const float*)d_in[0];   // (L,B,64)
  const float* xa  = (const float*)d_in[1];   // (L,B,32)
  const float* W_r = (const float*)d_in[2];   // (16384,160)
  const float* b_r = (const float*)d_in[3];   // (16384)
  const float* W_i = (const float*)d_in[4];   // (8192,160)
  const float* b_i = (const float*)d_in[5];   // (8192)
  const float* W_o = (const float*)d_in[6];   // (128,160)
  const float* b_o = (const float*)d_in[7];   // (128)

  float* out_h = (float*)d_out;                                   // hs
  float* out_c = (float*)d_out + (size_t)L_STEPS * BATCH * OUT_DIM; // cs

  // workspace layout (bytes)
  const int    WR_ELEMS = OUT_DIM * OUT_DIM * GATE_DIM; // 2,621,440
  const int    WI_ELEMS = IN_DIM * OUT_DIM * GATE_DIM;  // 1,310,720
  char* ws = (char*)d_ws;
  size_t off = 0;
  bf16_t* Wr_bf   = (bf16_t*)(ws + off); off += (size_t)WR_ELEMS * 2;
  bf16_t* Wi_bf   = (bf16_t*)(ws + off); off += (size_t)WI_ELEMS * 2;
  float*  c_state = (float*)(ws + off);  off += (size_t)BATCH * OUT_DIM * 4;
  float*  c_acc   = (float*)(ws + off);  off += (size_t)BATCH * OUT_DIM * 4;
  bf16_t* g_bf    = (bf16_t*)(ws + off); off += (size_t)BATCH * GATE_DIM * 2;
  float*  o_gate  = (float*)(ws + off);  off += (size_t)BATCH * OUT_DIM * 4;
  // total ~8.34 MB

  // 1) weights -> bf16 (L2-resident for the whole scan)
  cvt_bf16_kernel<<<(WR_ELEMS + 255) / 256, 256, 0, stream>>>(W_r, Wr_bf, WR_ELEMS);
  cvt_bf16_kernel<<<(WI_ELEMS + 255) / 256, 256, 0, stream>>>(W_i, Wi_bf, WI_ELEMS);

  // 2) initial carry / gate prep for t = 0
  state_kernel<<<BATCH, OUT_DIM, 0, stream>>>(xa, W_o, b_o, c_state, c_acc,
                                              g_bf, o_gate, out_h, out_c, -1);

  // 3) scan
  for (int t = 0; t < L_STEPS; ++t) {
    gate_gemm_kernel<OUT_DIM, true><<<dim3(OUT_DIM, BATCH / 16), 128, 0, stream>>>(
        g_bf, Wr_bf, b_r, c_state, xm, c_acc, t);
    gate_gemm_kernel<IN_DIM, false><<<dim3(IN_DIM, BATCH / 16), 128, 0, stream>>>(
        g_bf, Wi_bf, b_i, c_state, xm, c_acc, t);
    state_kernel<<<BATCH, OUT_DIM, 0, stream>>>(xa, W_o, b_o, c_state, c_acc,
                                                g_bf, o_gate, out_h, out_c, t);
  }
}
